// GraphAttention_42228118454687
// MI455X (gfx1250) — matrix operand order repaired and checked by reading, not compiled, not run on hardware
//
#include <hip/hip_runtime.h>

// CDNA5 / gfx1250: wave32, WMMA 16x16x32 f16 -> f32 accumulate.
typedef __attribute__((ext_vector_type(16))) _Float16 v16h;
typedef __attribute__((ext_vector_type(8)))  float    v8f;

#define NN     32
#define CDIM   64
#define ODIM   64
#define TDIM   512
#define VDIM   25
#define TB     4                      // t's per workgroup (one per wave32)
#define NEG_BIG (-9000000000000000.0f)
#define ALPHA   0.2f

// Per-t LDS arena layout (units: halves). Pitches chosen so WMMA fragment
// reads are 16B-aligned and spread across the 64 LDS banks.
#define HA_PAD   72                    // hA: 32 rows (v) x 64 (c), pitch 72
#define ATT_PAD  40                    // attention: 32 x 32, pitch 40
#define WHT_PAD  40                    // Wh^T (f16): 64 rows (o) x 32 (v), pitch 40
#define HA_OFF   0
#define ATT_OFF  (32 * HA_PAD)                  // 2304
#define WHT_OFF  (ATT_OFF + 32 * ATT_PAD)       // 3584
#define ARENA_H  (WHT_OFF + 64 * WHT_PAD)       // 6144 halves = 12288 B
#define STG_PITCH 65                   // f32 h' staging pitch (bank-conflict-free)

__device__ __forceinline__ v16h ldfrag(const _Float16* p0, const _Float16* p1) {
  union { v16h v; uint4 q[2]; } u;
  u.q[0] = *reinterpret_cast<const uint4*>(p0);
  u.q[1] = *reinterpret_cast<const uint4*>(p1);
  return u.v;
}

__global__ __launch_bounds__(128)
void gat_fused(const float* __restrict__ h, const int* __restrict__ adj,
               const float* __restrict__ W, const float* __restrict__ a,
               float* __restrict__ out)
{
  __shared__ __align__(16) _Float16 s_arena[TB][ARENA_H];   // 49152 B
  __shared__ __align__(16) _Float16 s_Wt[64 * HA_PAD];      // W^T f16, 9216 B
  __shared__ float    s_wa[128];                            // W.a1 | W.a2
  __shared__ float    s_f2[TB][32];
  __shared__ unsigned s_adjm[96];                           // bitmask per (k,u)

  const int tid  = threadIdx.x;
  const int wv   = tid >> 5;          // wave id == local t index
  const int lane = tid & 31;
  const int l15  = lane & 15;
  const int hi   = lane >> 4;         // half-wave select
  const int koff = hi * 8;            // A-fragment K offset (ISA 16-bit A layout)
  const int hi16 = hi * 8;            // B-fragment K offset

  const int n  = blockIdx.x >> 7;                 // T/TB = 128
  const int t0 = (blockIdx.x & 127) * TB;
  const size_t hBase = (size_t)n * CDIM * TDIM * VDIM;

  // ---- stage W^T as f16 (B-matrix source for matmul1) ----
  for (int i = tid; i < 64 * 64; i += 128) {
    int c = i >> 6, o = i & 63;
    s_Wt[o * HA_PAD + c] = (_Float16)W[i];
  }
  // ---- wa1/wa2: fold attention vectors through W  (f = h^T . (W a)) ----
  {
    int hv = tid >> 6, c = tid & 63;
    float s = 0.f;
    #pragma unroll 8
    for (int o = 0; o < 64; ++o) s += W[c * 64 + o] * a[hv * 64 + o];
    s_wa[tid] = s;
  }
  // ---- adjacency -> 25-bit masks per (k, u) ----
  if (tid < 96) {
    int k = tid >> 5, u = tid & 31;
    unsigned m = 0;
    if (u < 25) {
      for (int v = 0; v < 25; ++v)
        if (adj[(k * 25 + u) * 25 + v] > 0) m |= (1u << v);
    }
    s_adjm[tid] = m;
  }
  // ---- stage h[n,:,t,:] tiles as f16 A-matrices (coalesced per c) ----
  for (int i = tid; i < CDIM * TB * VDIM; i += 128) {          // 6400
    int c  = i / (TB * VDIM);
    int j  = i - c * (TB * VDIM);
    int tl = j / VDIM, v = j - tl * VDIM;
    float hv = h[hBase + (size_t)c * (TDIM * VDIM) + (size_t)t0 * VDIM + j];
    s_arena[tl][HA_OFF + v * HA_PAD + c] = (_Float16)hv;
  }
  // zero the padded rows v=25..31 (stale LDS bits could be f16 NaN/Inf)
  for (int i = tid; i < TB * 7 * 64; i += 128) {
    int tl = i / 448, r = i - tl * 448;
    s_arena[tl][HA_OFF + (25 + (r >> 6)) * HA_PAD + (r & 63)] = (_Float16)0.f;
  }
  __syncthreads();

  _Float16* hA   = &s_arena[wv][HA_OFF];
  _Float16* attA = &s_arena[wv][ATT_OFF];
  _Float16* WhT  = &s_arena[wv][WHT_OFF];

  // ================= matmul1: Wh(32x64) = hA(32x64[K]) x W(64x64) =========
  v16h af[2][2];
  #pragma unroll
  for (int mt = 0; mt < 2; ++mt) {
    #pragma unroll
    for (int kb = 0; kb < 2; ++kb) {
      const _Float16* p = hA + (mt * 16 + l15) * HA_PAD + kb * 32 + koff;
      af[mt][kb] = ldfrag(p, p + 16);
    }
  }
  v16h bf[4][2];
  #pragma unroll
  for (int nt = 0; nt < 4; ++nt) {
    #pragma unroll
    for (int kb = 0; kb < 2; ++kb) {
      const _Float16* p = s_Wt + (nt * 16 + l15) * HA_PAD + kb * 32 + hi16;
      bf[nt][kb] = ldfrag(p, p + 16);
    }
  }
  #pragma unroll
  for (int mt = 0; mt < 2; ++mt) {
    #pragma unroll
    for (int nt = 0; nt < 4; ++nt) {
      v8f acc = {};
      acc = __builtin_amdgcn_wmma_f32_16x16x32_f16(false, af[mt][0], false, bf[nt][0],
                                                   (short)0, acc, false, false);
      acc = __builtin_amdgcn_wmma_f32_16x16x32_f16(false, af[mt][1], false, bf[nt][1],
                                                   (short)0, acc, false, false);
      #pragma unroll
      for (int r = 0; r < 8; ++r) {     // D layout: row = r + 8*hi, col = l15
        WhT[(nt * 16 + l15) * WHT_PAD + (mt * 16 + r + 8 * hi)] = (_Float16)acc[r];
      }
    }
  }

  // ================= f1/f2 per vertex (lane = v) ===========================
  float f1 = 0.f, f2 = 0.f;
  #pragma unroll 8
  for (int c = 0; c < 64; ++c) {
    float hv = (float)hA[lane * HA_PAD + c];
    f1 += hv * s_wa[c];
    f2 += hv * s_wa[64 + c];
  }
  s_f2[wv][lane] = f2;                 // same-wave LDS: DS ops are in-order

  // ================= masked 3-way softmax, summed ==========================
  float att[25];
  #pragma unroll
  for (int v = 0; v < 25; ++v) att[v] = 0.f;
  if (lane < 25) {
    #pragma unroll
    for (int k = 0; k < 3; ++k) {
      unsigned msk = s_adjm[k * 32 + lane];
      float sc[25], mx = NEG_BIG;
      #pragma unroll
      for (int v = 0; v < 25; ++v) {
        float x = f1 + s_f2[wv][v];
        x = (x >= 0.f) ? x : ALPHA * x;                 // leaky_relu
        sc[v] = ((msk >> v) & 1u) ? x : NEG_BIG;
        mx = fmaxf(mx, sc[v]);
      }
      float se = 0.f;
      #pragma unroll
      for (int v = 0; v < 25; ++v) { sc[v] = __expf(sc[v] - mx); se += sc[v]; }
      float inv = 1.0f / se;
      #pragma unroll
      for (int v = 0; v < 25; ++v) att[v] += sc[v] * inv;
    }
  }
  {
    _Float16* row = attA + lane * ATT_PAD;   // all 32 lanes write (zero pads)
    #pragma unroll
    for (int v = 0; v < 25; ++v) row[v] = (lane < 25) ? (_Float16)att[v] : (_Float16)0.f;
    #pragma unroll
    for (int v = 25; v < 32; ++v) row[v] = (_Float16)0.f;
  }

  // ================= matmul2: h'(32x64) = att(32x32[K]) x Wh(32x64) ========
  v16h a2[2];
  #pragma unroll
  for (int mt = 0; mt < 2; ++mt) {
    const _Float16* p = attA + (mt * 16 + l15) * ATT_PAD + koff;
    a2[mt] = ldfrag(p, p + 16);
  }
  v16h b2[4];
  #pragma unroll
  for (int nt = 0; nt < 4; ++nt) {
    const _Float16* p = WhT + (nt * 16 + l15) * WHT_PAD + hi16;
    b2[nt] = ldfrag(p, p + 16);
  }
  v8f hp[2][4];                         // keep all results in regs before the
  #pragma unroll                        // staging stores alias this arena
  for (int mt = 0; mt < 2; ++mt) {
    #pragma unroll
    for (int nt = 0; nt < 4; ++nt) {
      v8f acc = {};
      hp[mt][nt] = __builtin_amdgcn_wmma_f32_16x16x32_f16(false, a2[mt], false, b2[nt],
                                                          (short)0, acc, false, false);
    }
  }
  float* stg = reinterpret_cast<float*>(&s_arena[wv][0]);   // reuse arena as f32
  #pragma unroll
  for (int mt = 0; mt < 2; ++mt) {
    #pragma unroll
    for (int nt = 0; nt < 4; ++nt) {
      #pragma unroll
      for (int r = 0; r < 8; ++r)
        stg[(mt * 16 + r + 8 * hi) * STG_PITCH + nt * 16 + l15] = hp[mt][nt][r];
    }
  }
  __syncthreads();

  // ================= ELU + coalesced transposed writeout ===================
  // out[n,o,t,u]: for this block, o-run of TB*VDIM = 100 contiguous floats.
  const size_t oBase = (size_t)n * ODIM * TDIM * VDIM + (size_t)t0 * VDIM;
  for (int o = wv; o < ODIM; o += TB) {
    for (int i = lane; i < TB * VDIM; i += 32) {
      int tl = i / 25, u = i - tl * 25;
      float x = reinterpret_cast<float*>(&s_arena[tl][0])[u * STG_PITCH + o];
      x = (x > 0.f) ? x : (__expf(x) - 1.f);              // elu
      out[oBase + (size_t)o * (TDIM * VDIM) + i] = x;
    }
  }
}

extern "C" void kernel_launch(void* const* d_in, const int* in_sizes, int n_in,
                              void* d_out, int out_size, void* d_ws, size_t ws_size,
                              hipStream_t stream) {
  (void)in_sizes; (void)n_in; (void)out_size; (void)d_ws; (void)ws_size;
  const float* h   = (const float*)d_in[0];
  const int*   adj = (const int*)d_in[1];
  const float* W   = (const float*)d_in[2];
  const float* a   = (const float*)d_in[3];
  float* out = (float*)d_out;
  dim3 grid(NN * (TDIM / TB));   // 4096 workgroups
  dim3 block(128);               // 4 wave32's, one t each
  gat_fused<<<grid, block, 0, stream>>>(h, adj, W, a, out);
}
